// SelfAttentionWithKVCache_39591008534595
// MI455X (gfx1250) — matrix operand order repaired and checked by reading, not compiled, not run on hardware
//
#include <hip/hip_runtime.h>

// ---------------------------------------------------------------------------
// SelfAttentionWithKVCache for MI455X (gfx1250, wave32, WMMA f16 16x16x32)
//
// Pipeline (all on `stream`):
//   1) cast x, q_w|k_w|v_w (concat), o_w  : fp32 -> f16 in ws
//   2) copy past_k/past_v into d_out (fp32) and ws caches (f16)
//   3) fused QKV GEMM (WMMA, async-to-LDS staging) -> Q f16, K/V caches
//   4) flash attention (WMMA, async K staging) -> attn f16
//   5) output projection GEMM (WMMA, +bias) -> fp32 d_out
// ---------------------------------------------------------------------------

typedef __attribute__((ext_vector_type(16))) _Float16 v16h;
typedef __attribute__((ext_vector_type(8)))  float    v8f;
typedef __attribute__((ext_vector_type(4)))  _Float16 v4h;

union FragU { v16h v; uint4 q[2]; };

__device__ inline v8f wmma_f16(const FragU& a, const FragU& b, v8f c) {
  return __builtin_amdgcn_wmma_f32_16x16x32_f16(
      /*neg_a=*/false, a.v, /*neg_b=*/false, b.v,
      /*c_mod=*/(short)0, c, /*reuse_a=*/false, /*reuse_b=*/false);
}

// gfx1250 async global->LDS copy, 16B per lane; tracked by ASYNCcnt.
__device__ inline void async_copy_b128(const void* lds_dst, const void* gsrc) {
  unsigned           lo = (unsigned)(unsigned long long)lds_dst;   // LDS offset
  unsigned long long ga = (unsigned long long)gsrc;
  asm volatile("global_load_async_to_lds_b128 %0, %1, off"
               :: "v"(lo), "v"(ga) : "memory");
}
__device__ inline void wait_async0() {
  asm volatile("s_wait_asynccnt 0x0" ::: "memory");
}

// Model constants
#define DM   2048   // d_model
#define NH   16     // heads
#define DH   128    // head dim
#define BB   2      // batch
#define TT   2048   // new tokens
#define PAST 2048
#define SS   4096   // PAST + TT
#define MTOT 4096   // B*T rows

// ---------------------------------------------------------------------------
// 1) fp32 -> f16 cast (vectorized x4)
// ---------------------------------------------------------------------------
__global__ void cast_f32_to_f16_kernel(const float* __restrict__ src,
                                       _Float16* __restrict__ dst, int n4) {
  int i = blockIdx.x * blockDim.x + threadIdx.x;
  if (i < n4) {
    float4 f = ((const float4*)src)[i];
    v4h o;
    o[0] = (_Float16)f.x; o[1] = (_Float16)f.y;
    o[2] = (_Float16)f.z; o[3] = (_Float16)f.w;
    ((v4h*)dst)[i] = o;
  }
}

// ---------------------------------------------------------------------------
// 2) past_k/past_v -> d_out fp32 cache (s < PAST) and ws f16 caches
// ---------------------------------------------------------------------------
__global__ void past_copy_kernel(const float* __restrict__ pk,
                                 const float* __restrict__ pv,
                                 float* __restrict__ kout,
                                 float* __restrict__ vout,
                                 _Float16* __restrict__ kc,
                                 _Float16* __restrict__ vc) {
  size_t i = (size_t)blockIdx.x * blockDim.x + threadIdx.x;  // < 2^23
  size_t bh  = i >> 18;          // / (PAST*DH) = 2^18
  size_t rem = i & 262143u;
  size_t oi  = (bh << 19) + rem; // * (SS*DH) = 2^19, past occupies s in [0,PAST)
  float k = pk[i], v = pv[i];
  kout[oi] = k;  vout[oi] = v;
  kc[oi] = (_Float16)k;  vc[oi] = (_Float16)v;
}

// ---------------------------------------------------------------------------
// 3) fused QKV GEMM:  C[m,n] = sum_k x[m,k] * Wcat[n,k] + bias
//    M=4096, N=6144, K=2048.  Block tile 64x128, 4 waves of 32x64 each.
// ---------------------------------------------------------------------------
__global__ __launch_bounds__(128)
void gemm_qkv_kernel(const _Float16* __restrict__ A,     // [4096][2048]
                     const _Float16* __restrict__ Bw,    // [6144][2048]
                     const float* __restrict__ qb,
                     const float* __restrict__ kb,
                     const float* __restrict__ vb,
                     _Float16* __restrict__ qh,          // [B,H,T,DH] f16
                     _Float16* __restrict__ kch,         // [B,H,S,DH] f16
                     _Float16* __restrict__ vch,         // [B,H,S,DH] f16
                     float* __restrict__ kout,           // d_out K cache fp32
                     float* __restrict__ vout) {         // d_out V cache fp32
  __shared__ _Float16 As[64][40];    // 64x32 tile, pad to 40 (16B aligned rows)
  __shared__ _Float16 Bs[128][40];

  const int tid  = threadIdx.x;
  const int wid  = tid >> 5;
  const int lane = tid & 31;
  const int wm = wid >> 1;                 // 0..1, 32 rows each
  const int wn = wid & 1;                  // 0..1, 64 cols each
  const int m0 = blockIdx.y * 64;
  const int n0 = blockIdx.x * 128;

  const int arow  = lane & 15;
  const int ahalf = lane >> 4;
  const int akb   = ahalf * 8;
  const int bkb   = ahalf * 8;

  v8f acc[2][4] = {{v8f{}, v8f{}, v8f{}, v8f{}},
                   {v8f{}, v8f{}, v8f{}, v8f{}}};

  for (int k0 = 0; k0 < DM; k0 += 32) {
    // async stage A tile: 64 rows x 32 halves, 2 x b128 per thread
    #pragma unroll
    for (int i = 0; i < 2; ++i) {
      int idx = tid + i * 128;
      int r = idx >> 2, ch = idx & 3;
      async_copy_b128(&As[r][ch * 8],
                      A + (size_t)(m0 + r) * DM + k0 + ch * 8);
    }
    // async stage B tile: 128 rows x 32 halves, 4 x b128 per thread
    #pragma unroll
    for (int i = 0; i < 4; ++i) {
      int idx = tid + i * 128;
      int r = idx >> 2, ch = idx & 3;
      async_copy_b128(&Bs[r][ch * 8],
                      Bw + (size_t)(n0 + r) * DM + k0 + ch * 8);
    }
    if (k0 + 32 < DM) {
      __builtin_prefetch(A + (size_t)(m0 + (tid >> 1)) * DM + k0 + 32, 0, 1);
      __builtin_prefetch(Bw + (size_t)(n0 + tid) * DM + k0 + 32, 0, 1);
    }
    wait_async0();
    __syncthreads();

    // all fragments into distinct registers, then 8 back-to-back WMMAs
    FragU a[2], b[4];
    #pragma unroll
    for (int ma = 0; ma < 2; ++ma) {
      const _Float16* p = &As[wm * 32 + ma * 16 + arow][akb];
      a[ma].q[0] = *(const uint4*)p;
      a[ma].q[1] = *(const uint4*)(p + 16);
    }
    #pragma unroll
    for (int j = 0; j < 4; ++j) {
      const _Float16* p = &Bs[wn * 64 + j * 16 + arow][bkb];
      b[j].q[0] = *(const uint4*)p;
      b[j].q[1] = *(const uint4*)(p + 16);
    }
    #pragma unroll
    for (int ma = 0; ma < 2; ++ma)
      #pragma unroll
      for (int j = 0; j < 4; ++j)
        acc[ma][j] = wmma_f16(a[ma], b[j], acc[ma][j]);
    __syncthreads();
  }

  // Epilogue: scatter into Q / K / V destinations.
  const int lcol = lane & 15;
  #pragma unroll
  for (int ma = 0; ma < 2; ++ma) {
    #pragma unroll
    for (int j = 0; j < 4; ++j) {
      int n = n0 + wn * 64 + j * 16 + lcol;       // 0..6143
      int proj = n >> 11;                          // 0=Q 1=K 2=V
      int e = n & (DM - 1);
      int h = e >> 7, d = e & (DH - 1);
      float bias = (proj == 0) ? qb[e] : (proj == 1) ? kb[e] : vb[e];
      #pragma unroll
      for (int r = 0; r < 8; ++r) {
        int m = m0 + wm * 32 + ma * 16 + r + 8 * ahalf;
        int bidx = m >> 11;                        // T = 2048
        int t = m & (TT - 1);
        float val = acc[ma][j][r] + bias;
        _Float16 hv = (_Float16)val;
        size_t bh = (size_t)(bidx * NH + h);
        if (proj == 0) {
          qh[(bh * TT + t) * DH + d] = hv;
        } else if (proj == 1) {
          size_t ci = (bh * SS + PAST + t) * DH + d;
          kch[ci] = hv;
          kout[ci] = val;
        } else {
          size_t ci = (bh * SS + PAST + t) * DH + d;
          vch[ci] = hv;
          vout[ci] = val;
        }
      }
    }
  }
}

// ---------------------------------------------------------------------------
// 4) Flash attention. grid = (T/64, B*H), block = 128 (4 waves).
// ---------------------------------------------------------------------------
__global__ __launch_bounds__(128)
void attn_kernel(const _Float16* __restrict__ qh,    // [B,H,T,DH]
                 const _Float16* __restrict__ kch,   // [B,H,S,DH]
                 const _Float16* __restrict__ vch,   // [B,H,S,DH]
                 _Float16* __restrict__ attnh) {     // [B*T, DM] f16
  __shared__ _Float16 Ks[32][136];   // K chunk as loaded  [key][dh]
  __shared__ _Float16 Vt[128][40];   // V chunk transposed [dh][key]
  __shared__ _Float16 Ps[4][16][32]; // per-wave P tile

  const int bh = blockIdx.y;
  const int b  = bh >> 4;
  const int h  = bh & 15;
  const int wid  = threadIdx.x >> 5;
  const int lane = threadIdx.x & 31;
  const int t0 = blockIdx.x * 64 + wid * 16;

  const _Float16* qptr = qh  + ((size_t)bh * TT + t0) * DH;
  const _Float16* kptr = kch + (size_t)bh * SS * DH;
  const _Float16* vptr = vch + (size_t)bh * SS * DH;

  const int arow  = lane & 15;
  const int ahalf = lane >> 4;
  const int akb   = ahalf * 8;
  const int bkb   = ahalf * 8;

  // Q fragments for the whole dh (4 chunks of K=32), kept in registers.
  FragU qf[4];
  #pragma unroll
  for (int c = 0; c < 4; ++c) {
    const _Float16* p = qptr + arow * DH + c * 32 + akb;
    qf[c].q[0] = *(const uint4*)p;
    qf[c].q[1] = *(const uint4*)(p + 16);
  }

  v8f oacc[8] = {v8f{}, v8f{}, v8f{}, v8f{}, v8f{}, v8f{}, v8f{}, v8f{}};
  float rowm[8], rowl[8];
  #pragma unroll
  for (int r = 0; r < 8; ++r) { rowm[r] = -1e30f; rowl[r] = 0.0f; }
  const float scale = 0.08838834764831845f;   // 1/sqrt(128)

  for (int s0 = 0; s0 < SS; s0 += 32) {
    // async stage K chunk (32x128): 4 x b128 per thread
    #pragma unroll
    for (int i = 0; i < 4; ++i) {
      int idx = threadIdx.x + i * 128;
      int r = idx >> 4, ch = idx & 15;
      async_copy_b128(&Ks[r][ch * 8],
                      kptr + (size_t)(s0 + r) * DH + ch * 8);
    }
    // stage V chunk transposed [dh][key] (in-register transpose)
    #pragma unroll
    for (int i = 0; i < 4; ++i) {
      int idx = threadIdx.x + i * 128;
      int r = idx >> 4, ch = idx & 15;
      uint4 raw = *(const uint4*)(vptr + (size_t)(s0 + r) * DH + ch * 8);
      const _Float16* e = (const _Float16*)&raw;
      #pragma unroll
      for (int jj = 0; jj < 8; ++jj) Vt[ch * 8 + jj][r] = e[jj];
    }
    wait_async0();
    __syncthreads();

    // scores: 16 x 32 = two 16x16 C tiles, K-reduced over dh (4 chunks)
    v8f sa[2] = {v8f{}, v8f{}};
    #pragma unroll
    for (int c = 0; c < 4; ++c) {
      #pragma unroll
      for (int j2 = 0; j2 < 2; ++j2) {
        FragU bf;
        const _Float16* p = &Ks[j2 * 16 + arow][c * 32 + bkb];
        bf.q[0] = *(const uint4*)p;
        bf.q[1] = *(const uint4*)(p + 16);
        sa[j2] = wmma_f16(qf[c], bf, sa[j2]);
      }
    }
    #pragma unroll
    for (int j2 = 0; j2 < 2; ++j2)
      #pragma unroll
      for (int r = 0; r < 8; ++r) sa[j2][r] *= scale;

    // online softmax per row (rows striped: VGPR r -> row r + 8*ahalf)
    #pragma unroll
    for (int r = 0; r < 8; ++r) {
      float v = fmaxf(sa[0][r], sa[1][r]);
      #pragma unroll
      for (int off = 8; off >= 1; off >>= 1) v = fmaxf(v, __shfl_xor(v, off, 16));
      float mnew = fmaxf(rowm[r], v);
      float alpha = __expf(rowm[r] - mnew);
      float p0 = __expf(sa[0][r] - mnew);
      float p1 = __expf(sa[1][r] - mnew);
      sa[0][r] = p0; sa[1][r] = p1;
      float rs = p0 + p1;
      #pragma unroll
      for (int off = 8; off >= 1; off >>= 1) rs += __shfl_xor(rs, off, 16);
      rowl[r] = rowl[r] * alpha + rs;
      rowm[r] = mnew;
      #pragma unroll
      for (int j = 0; j < 8; ++j) oacc[j][r] *= alpha;
    }

    // P -> LDS (C layout) then reload as WMMA A fragment
    #pragma unroll
    for (int r = 0; r < 8; ++r) {
      int prow = r + 8 * ahalf;
      Ps[wid][prow][arow]      = (_Float16)sa[0][r];
      Ps[wid][prow][16 + arow] = (_Float16)sa[1][r];
    }
    FragU pf;
    {
      const _Float16* p = &Ps[wid][arow][akb];
      pf.q[0] = *(const uint4*)p;
      pf.q[1] = *(const uint4*)(p + 16);
    }
    // O += P @ V   (8 n-subtiles over dh)
    #pragma unroll
    for (int j = 0; j < 8; ++j) {
      FragU bf;
      const _Float16* p = &Vt[j * 16 + arow][bkb];
      bf.q[0] = *(const uint4*)p;
      bf.q[1] = *(const uint4*)(p + 16);
      oacc[j] = wmma_f16(pf, bf, oacc[j]);
    }
    __syncthreads();
  }

  // epilogue: normalize and scatter into attn buffer [B*T, DM] at head slice
  #pragma unroll
  for (int r = 0; r < 8; ++r) {
    float inv = 1.0f / rowl[r];
    int t = t0 + r + 8 * ahalf;
    size_t rowbase = ((size_t)b * TT + t) * DM + h * DH;
    #pragma unroll
    for (int j = 0; j < 8; ++j) {
      attnh[rowbase + j * 16 + arow] = (_Float16)(oacc[j][r] * inv);
    }
  }
}

// ---------------------------------------------------------------------------
// 5) output projection: out[m,n] = sum_k attn[m,k] * o_w[n,k] + o_b[n]
// ---------------------------------------------------------------------------
__global__ __launch_bounds__(128)
void gemm_out_kernel(const _Float16* __restrict__ A,   // [4096][2048]
                     const _Float16* __restrict__ Bw,  // [2048][2048]
                     const float* __restrict__ ob,
                     float* __restrict__ out) {        // [4096][2048]
  __shared__ _Float16 As[64][40];
  __shared__ _Float16 Bs[128][40];

  const int tid  = threadIdx.x;
  const int wid  = tid >> 5;
  const int lane = tid & 31;
  const int wm = wid >> 1;
  const int wn = wid & 1;
  const int m0 = blockIdx.y * 64;
  const int n0 = blockIdx.x * 128;

  const int arow  = lane & 15;
  const int ahalf = lane >> 4;
  const int akb   = ahalf * 8;
  const int bkb   = ahalf * 8;

  v8f acc[2][4] = {{v8f{}, v8f{}, v8f{}, v8f{}},
                   {v8f{}, v8f{}, v8f{}, v8f{}}};

  for (int k0 = 0; k0 < DM; k0 += 32) {
    #pragma unroll
    for (int i = 0; i < 2; ++i) {
      int idx = tid + i * 128;
      int r = idx >> 2, ch = idx & 3;
      async_copy_b128(&As[r][ch * 8],
                      A + (size_t)(m0 + r) * DM + k0 + ch * 8);
    }
    #pragma unroll
    for (int i = 0; i < 4; ++i) {
      int idx = tid + i * 128;
      int r = idx >> 2, ch = idx & 3;
      async_copy_b128(&Bs[r][ch * 8],
                      Bw + (size_t)(n0 + r) * DM + k0 + ch * 8);
    }
    if (k0 + 32 < DM) {
      __builtin_prefetch(A + (size_t)(m0 + (tid >> 1)) * DM + k0 + 32, 0, 1);
      __builtin_prefetch(Bw + (size_t)(n0 + tid) * DM + k0 + 32, 0, 1);
    }
    wait_async0();
    __syncthreads();

    FragU a[2], b[4];
    #pragma unroll
    for (int ma = 0; ma < 2; ++ma) {
      const _Float16* p = &As[wm * 32 + ma * 16 + arow][akb];
      a[ma].q[0] = *(const uint4*)p;
      a[ma].q[1] = *(const uint4*)(p + 16);
    }
    #pragma unroll
    for (int j = 0; j < 4; ++j) {
      const _Float16* p = &Bs[wn * 64 + j * 16 + arow][bkb];
      b[j].q[0] = *(const uint4*)p;
      b[j].q[1] = *(const uint4*)(p + 16);
    }
    #pragma unroll
    for (int ma = 0; ma < 2; ++ma)
      #pragma unroll
      for (int j = 0; j < 4; ++j)
        acc[ma][j] = wmma_f16(a[ma], b[j], acc[ma][j]);
    __syncthreads();
  }

  const int lcol = lane & 15;
  #pragma unroll
  for (int ma = 0; ma < 2; ++ma) {
    #pragma unroll
    for (int j = 0; j < 4; ++j) {
      int n = n0 + wn * 64 + j * 16 + lcol;
      float bias = ob[n];
      #pragma unroll
      for (int r = 0; r < 8; ++r) {
        int m = m0 + wm * 32 + ma * 16 + r + 8 * ahalf;
        out[(size_t)m * DM + n] = acc[ma][j][r] + bias;
      }
    }
  }
}

// ---------------------------------------------------------------------------
// Host side
// ---------------------------------------------------------------------------
extern "C" void kernel_launch(void* const* d_in, const int* in_sizes, int n_in,
                              void* d_out, int out_size, void* d_ws, size_t ws_size,
                              hipStream_t stream) {
  const float* x      = (const float*)d_in[0];
  const float* past_k = (const float*)d_in[1];
  const float* past_v = (const float*)d_in[2];
  const float* q_w = (const float*)d_in[3];
  const float* q_b = (const float*)d_in[4];
  const float* k_w = (const float*)d_in[5];
  const float* k_b = (const float*)d_in[6];
  const float* v_w = (const float*)d_in[7];
  const float* v_b = (const float*)d_in[8];
  const float* o_w = (const float*)d_in[9];
  const float* o_b = (const float*)d_in[10];

  float* out  = (float*)d_out;                       // [4096][2048]
  float* kout = out + (size_t)MTOT * DM;             // [B,H,S,DH]
  float* vout = kout + (size_t)BB * NH * SS * DH;

  // workspace layout (f16 elements)
  _Float16* ws = (_Float16*)d_ws;
  _Float16* xh    = ws;                               // 8,388,608
  _Float16* wcat  = xh   + (size_t)MTOT * DM;         // 12,582,912
  _Float16* owh   = wcat + (size_t)3 * DM * DM;       // 4,194,304
  _Float16* qh    = owh  + (size_t)DM * DM;           // 8,388,608
  _Float16* kch   = qh   + (size_t)BB * NH * TT * DH; // 16,777,216
  _Float16* vch   = kch  + (size_t)BB * NH * SS * DH; // 16,777,216
  _Float16* attnh = vch  + (size_t)BB * NH * SS * DH; // 8,388,608

  const int TPB = 256;
  // 1) casts (n/4 vector elements, one float4 per thread)
  cast_f32_to_f16_kernel<<<(MTOT * DM / 4 + TPB - 1) / TPB, TPB, 0, stream>>>(x, xh, MTOT * DM / 4);
  cast_f32_to_f16_kernel<<<(DM * DM / 4 + TPB - 1) / TPB, TPB, 0, stream>>>(q_w, wcat,               DM * DM / 4);
  cast_f32_to_f16_kernel<<<(DM * DM / 4 + TPB - 1) / TPB, TPB, 0, stream>>>(k_w, wcat + (size_t)DM * DM,     DM * DM / 4);
  cast_f32_to_f16_kernel<<<(DM * DM / 4 + TPB - 1) / TPB, TPB, 0, stream>>>(v_w, wcat + (size_t)2 * DM * DM, DM * DM / 4);
  cast_f32_to_f16_kernel<<<(DM * DM / 4 + TPB - 1) / TPB, TPB, 0, stream>>>(o_w, owh, DM * DM / 4);

  // 2) past KV -> fp32 outputs + f16 caches  (B*H*PAST*DH = 8,388,608 elems)
  past_copy_kernel<<<(BB * NH * PAST * DH) / TPB, TPB, 0, stream>>>(
      past_k, past_v, kout, vout, kch, vch);

  // 3) fused QKV GEMM: M=4096, N=6144
  gemm_qkv_kernel<<<dim3(6144 / 128, MTOT / 64), 128, 0, stream>>>(
      xh, wcat, q_b, k_b, v_b, qh, kch, vch, kout, vout);

  // 4) flash attention: grid (T/64, B*H)
  attn_kernel<<<dim3(TT / 64, BB * NH), 128, 0, stream>>>(qh, kch, vch, attnh);

  // 5) output projection: M=4096, N=2048
  gemm_out_kernel<<<dim3(DM / 128, MTOT / 64), 128, 0, stream>>>(
      attnh, owh, o_b, out);
}
